// MultiHeadAttention_37349035606665
// MI455X (gfx1250) — matrix operand order repaired and checked by reading, not compiled, not run on hardware
//
#include <hip/hip_runtime.h>
#include <hip/hip_bf16.h>

// ---------------------------------------------------------------------------
// MultiHeadAttention forward for MI455X (gfx1250, wave32, WMMA).
//
// B=16, N=1024, C=1024, H=16, D=64. ~206 GFLOP vs ~330MB HBM -> compute
// bound -> all GEMMs on v_wmma_f32_16x16x32_f16 (f16 in, f32 accum).
// K loops are ping-pong unrolled (two named fragment sets, no rotation
// copies) so WMMAs overlap global loads with partial s_wait_loadcnt and
// zero v_mov buffer shuffling.
// ---------------------------------------------------------------------------

#define DIMC   1024
#define NHEADS 16
#define HDIM   64
#define NTOK   1024           // GRID*GRID
#define BATCH  16
#define NROWS  (BATCH * NTOK) // 16384

typedef _Float16 f16;
typedef __attribute__((ext_vector_type(16))) _Float16 v16h;
typedef __attribute__((ext_vector_type(8)))  _Float16 v8h;
typedef __attribute__((ext_vector_type(4)))  _Float16 v4h;
typedef __attribute__((ext_vector_type(8)))  float    v8f;
typedef __attribute__((ext_vector_type(4)))  float    v4f;

// ---- WMMA helpers ---------------------------------------------------------

static __device__ __forceinline__ v8f wmma_f16(v16h a, v16h b, v8f c) {
  return __builtin_amdgcn_wmma_f32_16x16x32_f16(
      false, a, false, b, (short)0, c, false, false);
}

// A-matrix 16x32 f16 fragment (ISA 7.12.2): lane r = lane%16 holds row r.
// halves j -> k = 16*hi + j (contiguous, matching B fragments).
static __device__ __forceinline__ v16h load_a_frag(const f16* row_base, int hi) {
  const v8h lo = *(const v8h*)(row_base + 16 * hi);
  const v8h hh = *(const v8h*)(row_base + 16 * hi + 8);
  v16h a;
#pragma unroll
  for (int j = 0; j < 8; ++j) { a[j] = lo[j]; a[j + 8] = hh[j]; }
  return a;
}

// B-matrix 32x16 f16 fragment: lane n = lane%16 holds column n,
// halves j -> k = 16*hi + j : one contiguous 32B run per lane.
static __device__ __forceinline__ v16h load_b_frag(const f16* p) {
  return *(const v16h*)p;
}

// ---- shared pipelined K loop: 32(M) x 64(N) wave tile, K = DIMC -----------
// Ping-pong buffers: half-step X consumes set X while loading set X^1 for
// k+32. No copy rotation -> no v_mov traffic in the steady state.

static __device__ __forceinline__ void gemm_k_loop(const f16* __restrict__ arow0,
                                                   const f16* __restrict__ arow1,
                                                   const f16* const brow[4],
                                                   int hi, v8f acc[2][4]) {
  v16h a0[2], b0[4], a1[2], b1[4];

  a0[0] = load_a_frag(arow0, hi);
  a0[1] = load_a_frag(arow1, hi);
#pragma unroll
  for (int f = 0; f < 4; ++f) b0[f] = load_b_frag(brow[f]);

  for (int k0 = 0; k0 < DIMC - 64; k0 += 64) {
    __builtin_prefetch(arow0 + k0 + 512, 0, 3);
    // load set1 <- k0+32; consume set0 (k0)
    a1[0] = load_a_frag(arow0 + k0 + 32, hi);
    a1[1] = load_a_frag(arow1 + k0 + 32, hi);
#pragma unroll
    for (int f = 0; f < 4; ++f) b1[f] = load_b_frag(brow[f] + k0 + 32);
#pragma unroll
    for (int f = 0; f < 4; ++f) {
      acc[0][f] = wmma_f16(a0[0], b0[f], acc[0][f]);
      acc[1][f] = wmma_f16(a0[1], b0[f], acc[1][f]);
    }
    // load set0 <- k0+64; consume set1 (k0+32)
    a0[0] = load_a_frag(arow0 + k0 + 64, hi);
    a0[1] = load_a_frag(arow1 + k0 + 64, hi);
#pragma unroll
    for (int f = 0; f < 4; ++f) b0[f] = load_b_frag(brow[f] + k0 + 64);
#pragma unroll
    for (int f = 0; f < 4; ++f) {
      acc[0][f] = wmma_f16(a1[0], b1[f], acc[0][f]);
      acc[1][f] = wmma_f16(a1[1], b1[f], acc[1][f]);
    }
  }
  // tail: set0 holds k = DIMC-64; load set1 <- DIMC-32, consume both
  a1[0] = load_a_frag(arow0 + DIMC - 32, hi);
  a1[1] = load_a_frag(arow1 + DIMC - 32, hi);
#pragma unroll
  for (int f = 0; f < 4; ++f) b1[f] = load_b_frag(brow[f] + DIMC - 32);
#pragma unroll
  for (int f = 0; f < 4; ++f) {
    acc[0][f] = wmma_f16(a0[0], b0[f], acc[0][f]);
    acc[1][f] = wmma_f16(a0[1], b0[f], acc[1][f]);
  }
#pragma unroll
  for (int f = 0; f < 4; ++f) {
    acc[0][f] = wmma_f16(a1[0], b1[f], acc[0][f]);
    acc[1][f] = wmma_f16(a1[1], b1[f], acc[1][f]);
  }
}

// ---- fp32 -> f16 converter ------------------------------------------------

__global__ void cvt_f32_to_f16_kernel(const float* __restrict__ src,
                                      f16* __restrict__ dst, int n4) {
  int idx = blockIdx.x * blockDim.x + threadIdx.x;
  for (; idx < n4; idx += gridDim.x * blockDim.x) {
    v4f s = *(const v4f*)(src + 4 * (size_t)idx);
    v4h d;
#pragma unroll
    for (int j = 0; j < 4; ++j) d[j] = (f16)s[j];
    *(v4h*)(dst + 4 * (size_t)idx) = d;
  }
}

// ---- Kernel 1: QKV GEMM + bias + RoPE -------------------------------------
// Block = 8 waves = 256 rows x 64 cols; a 64-col strip = one (q/k/v, head).
// q,k -> RoPE'd, stored f16 [B,H,N,D]; v stored transposed f16 [B,H,D,N].

__global__ __launch_bounds__(256, 1)
void qkv_rope_kernel(const f16* __restrict__ xh,        // [NROWS][DIMC]
                     const f16* __restrict__ wh,        // [3*DIMC][DIMC]
                     const float* __restrict__ bias,    // [3*DIMC]
                     f16* __restrict__ qb,              // [B,H,N,D]
                     f16* __restrict__ kb,              // [B,H,N,D]
                     f16* __restrict__ vt) {            // [B,H,D,N]
  const int lane = threadIdx.x & 31;
  const int wave = threadIdx.x >> 5;
  const int ln = lane & 15;
  const int hi = lane >> 4;
  const int row0 = blockIdx.y * 256 + wave * 32;
  const int col0 = blockIdx.x * 64;

  const f16* arow0 = xh + (size_t)(row0 + ln) * DIMC;
  const f16* arow1 = arow0 + (size_t)16 * DIMC;
  const f16* brow[4];
#pragma unroll
  for (int f = 0; f < 4; ++f)
    brow[f] = wh + (size_t)(col0 + 16 * f + ln) * DIMC + 16 * hi;

  v8f acc[2][4] = {};
  gemm_k_loop(arow0, arow1, brow, hi, acc);

#pragma unroll
  for (int f = 0; f < 4; ++f) {
    const float bv = bias[col0 + 16 * f + ln];
#pragma unroll
    for (int r = 0; r < 2; ++r)
#pragma unroll
      for (int i = 0; i < 8; ++i) acc[r][f][i] += bv;
  }

  const int sel = col0 >> 10;          // 0=q 1=k 2=v
  const int h   = (col0 & 1023) >> 6;
  const int b   = row0 >> 10;

  if (sel < 2) {
    // RoPE: pairs (d, d+32) = frags (fi, fi+2) of the same lane; freq = ln.
    const float invf = __expf(-(float)ln * 0.28782313662425572f); // ln(1e4)/32
    f16* dst = (sel == 0 ? qb : kb) + (size_t)(b * NHEADS + h) * NTOK * HDIM;
#pragma unroll
    for (int r = 0; r < 2; ++r) {
      const int n0 = (row0 & 1023) + 16 * r + 8 * hi;
#pragma unroll
      for (int i = 0; i < 8; ++i) {
        const int nt = n0 + i;
        const float py = (float)(nt >> 5);
        const float px = (float)(nt & 31);
#pragma unroll
        for (int fi = 0; fi < 2; ++fi) {
          const float ang = (fi ? px : py) * invf;
          const float c = __cosf(ang), s = __sinf(ang);
          const float t1 = acc[r][fi][i], t2 = acc[r][fi + 2][i];
          acc[r][fi][i]     = t1 * c - t2 * s;
          acc[r][fi + 2][i] = t1 * s + t2 * c;
        }
      }
#pragma unroll
      for (int f = 0; f < 4; ++f) {
        const int d = 16 * f + ln;
#pragma unroll
        for (int i = 0; i < 8; ++i)
          dst[(size_t)(n0 + i) * HDIM + d] = (f16)acc[r][f][i];
      }
    }
  } else {
    f16* dst = vt + (size_t)(b * NHEADS + h) * HDIM * NTOK;
#pragma unroll
    for (int r = 0; r < 2; ++r) {
      const int n0 = (row0 & 1023) + 16 * r + 8 * hi;
#pragma unroll
      for (int f = 0; f < 4; ++f) {
        const int d = 16 * f + ln;
#pragma unroll
        for (int i = 0; i < 8; ++i)
          dst[(size_t)d * NTOK + (n0 + i)] = (f16)acc[r][f][i];
      }
    }
  }
}

// ---- Kernel 2: flash attention --------------------------------------------
// Per wave: 16 q rows, online softmax over 32-wide kv chunks, kv loop
// ping-pong unrolled by 2 so next-chunk K fragments load during the softmax
// VALU work. P goes D-layout -> A-layout through a per-wave LDS tile
// (DS ops are in-order per wave; s_wait_dscnt 0 fences the transpose).

__global__ __launch_bounds__(256, 1)
void attn_kernel(const f16* __restrict__ qg,   // [B,H,N,D]
                 const f16* __restrict__ kg,   // [B,H,N,D]
                 const f16* __restrict__ vtg,  // [B,H,D,N]
                 f16* __restrict__ ao) {       // [B,N,C] f16
  __shared__ f16 lds_p[8 * 16 * 32];           // 1KB per wave

  const int lane = threadIdx.x & 31;
  const int wave = threadIdx.x >> 5;
  const int ln = lane & 15;
  const int hi = lane >> 4;

  const int bh   = blockIdx.x >> 3;
  const int qblk = blockIdx.x & 7;
  const int b = bh >> 4, h = bh & 15;
  const int nq0 = qblk * 128 + wave * 16;

  const f16* qrow = qg + ((size_t)bh * NTOK + nq0 + ln) * HDIM;
  const v16h qa0 = load_a_frag(qrow, hi);       // d 0..31
  const v16h qa1 = load_a_frag(qrow + 32, hi);  // d 32..63

  const f16* kbase = kg + (size_t)bh * NTOK * HDIM;
  const f16* vbase = vtg + (size_t)bh * HDIM * NTOK;
  f16* pt = lds_p + wave * 512;

  v8f o[4] = {};
  float m_r[8], l_r[8];
#pragma unroll
  for (int i = 0; i < 8; ++i) { m_r[i] = -1e30f; l_r[i] = 0.0f; }

  const float scale = 0.125f;

  auto load_k = [&](int nk, v16h (&kf)[2][2]) {
#pragma unroll
    for (int t = 0; t < 2; ++t) {
      const f16* kr = kbase + (size_t)(nk + 16 * t + ln) * HDIM + 16 * hi;
      kf[t][0] = load_b_frag(kr);
      kf[t][1] = load_b_frag(kr + 32);
    }
  };

  // One 32-wide kv chunk: S-WMMAs, prefetch of the next chunk's K frags,
  // V loads, online softmax, LDS transpose, P*V WMMAs.
  auto chunk = [&](int nk, v16h (&kcur)[2][2], v16h (&knxt)[2][2],
                   int nknxt, bool do_load) {
    v8f s[2] = {};
#pragma unroll
    for (int t = 0; t < 2; ++t) {
      s[t] = wmma_f16(qa0, kcur[t][0], s[t]);
      s[t] = wmma_f16(qa1, kcur[t][1], s[t]);
    }
    if (do_load) load_k(nknxt, knxt);
    v16h vb[4];
#pragma unroll
    for (int f = 0; f < 4; ++f)
      vb[f] = load_b_frag(vbase + (size_t)(16 * f + ln) * NTOK + nk + 16 * hi);

#pragma unroll
    for (int i = 0; i < 8; ++i) {
      const float s0 = s[0][i] * scale;
      const float s1 = s[1][i] * scale;
      float mx = fmaxf(s0, s1);
      mx = fmaxf(mx, __shfl_xor(mx, 1, 32));
      mx = fmaxf(mx, __shfl_xor(mx, 2, 32));
      mx = fmaxf(mx, __shfl_xor(mx, 4, 32));
      mx = fmaxf(mx, __shfl_xor(mx, 8, 32));
      const float mn = fmaxf(m_r[i], mx);
      const float alpha = __expf(m_r[i] - mn);
      const float p0 = __expf(s0 - mn);
      const float p1 = __expf(s1 - mn);
      float rs = p0 + p1;
      rs += __shfl_xor(rs, 1, 32);
      rs += __shfl_xor(rs, 2, 32);
      rs += __shfl_xor(rs, 4, 32);
      rs += __shfl_xor(rs, 8, 32);
      l_r[i] = l_r[i] * alpha + rs;
      m_r[i] = mn;
#pragma unroll
      for (int f = 0; f < 4; ++f) o[f][i] *= alpha;
      pt[(i + 8 * hi) * 32 + ln]      = (f16)p0;
      pt[(i + 8 * hi) * 32 + 16 + ln] = (f16)p1;
    }

    asm volatile("s_wait_dscnt 0" ::: "memory");
    const v16h pa = load_a_frag(pt + ln * 32, hi);
#pragma unroll
    for (int f = 0; f < 4; ++f) o[f] = wmma_f16(pa, vb[f], o[f]);
    asm volatile("" ::: "memory");  // keep next-iter LDS stores after loads
  };

  v16h ka[2][2], kbf[2][2];
  load_k(0, ka);
  for (int nk = 0; nk < NTOK - 64; nk += 64) {
    chunk(nk,      ka,  kbf, nk + 32, true);
    chunk(nk + 32, kbf, ka,  nk + 64, true);
  }
  chunk(NTOK - 64, ka,  kbf, NTOK - 32, true);
  chunk(NTOK - 32, kbf, ka,  0,         false);

#pragma unroll
  for (int f = 0; f < 4; ++f) {
    const int d = 16 * f + ln;
#pragma unroll
    for (int i = 0; i < 8; ++i) {
      const float val = o[f][i] * (1.0f / l_r[i]);
      ao[((size_t)b * NTOK + nq0 + i + 8 * hi) * DIMC + h * HDIM + d] = (f16)val;
    }
  }
}

// ---- Kernel 3: output projection GEMM + bias (fp32 out) -------------------

__global__ __launch_bounds__(256, 1)
void proj_kernel(const f16* __restrict__ ao,      // [NROWS][DIMC]
                 const f16* __restrict__ pwh,     // [DIMC][DIMC]
                 const float* __restrict__ pb,    // [DIMC]
                 float* __restrict__ out) {       // [NROWS][DIMC] fp32
  const int lane = threadIdx.x & 31;
  const int wave = threadIdx.x >> 5;
  const int ln = lane & 15;
  const int hi = lane >> 4;
  const int row0 = blockIdx.y * 256 + wave * 32;
  const int col0 = blockIdx.x * 64;

  const f16* arow0 = ao + (size_t)(row0 + ln) * DIMC;
  const f16* arow1 = arow0 + (size_t)16 * DIMC;
  const f16* brow[4];
#pragma unroll
  for (int f = 0; f < 4; ++f)
    brow[f] = pwh + (size_t)(col0 + 16 * f + ln) * DIMC + 16 * hi;

  v8f acc[2][4] = {};
  gemm_k_loop(arow0, arow1, brow, hi, acc);

#pragma unroll
  for (int f = 0; f < 4; ++f) {
    const float bv = pb[col0 + 16 * f + ln];
    const int c = col0 + 16 * f + ln;
#pragma unroll
    for (int r = 0; r < 2; ++r)
#pragma unroll
      for (int i = 0; i < 8; ++i)
        out[(size_t)(row0 + 16 * r + i + 8 * hi) * DIMC + c] = acc[r][f][i] + bv;
  }
}

// ---- launcher -------------------------------------------------------------

extern "C" void kernel_launch(void* const* d_in, const int* in_sizes, int n_in,
                              void* d_out, int out_size, void* d_ws, size_t ws_size,
                              hipStream_t stream) {
  const float* x      = (const float*)d_in[0];
  const float* qkv_w  = (const float*)d_in[1];
  const float* qkv_b  = (const float*)d_in[2];
  const float* proj_w = (const float*)d_in[3];
  const float* proj_b = (const float*)d_in[4];
  float* out = (float*)d_out;

  char* ws = (char*)d_ws;
  const size_t SZ_X  = (size_t)NROWS * DIMC * sizeof(f16);
  const size_t SZ_QW = (size_t)3 * DIMC * DIMC * sizeof(f16);
  const size_t SZ_PW = (size_t)DIMC * DIMC * sizeof(f16);
  const size_t SZ_T  = (size_t)BATCH * NHEADS * NTOK * HDIM * sizeof(f16);

  f16* xh  = (f16*)(ws);
  f16* qwh = (f16*)(ws + SZ_X);
  f16* pwh = (f16*)(ws + SZ_X + SZ_QW);
  f16* qbuf= (f16*)(ws + SZ_X + SZ_QW + SZ_PW);
  f16* kbuf= (f16*)(ws + SZ_X + SZ_QW + SZ_PW + SZ_T);
  f16* vt  = (f16*)(ws + SZ_X + SZ_QW + SZ_PW + 2 * SZ_T);
  f16* ao  = (f16*)(ws + SZ_X + SZ_QW + SZ_PW + 3 * SZ_T);

  cvt_f32_to_f16_kernel<<<4096, 256, 0, stream>>>(x, xh, NROWS * DIMC / 4);
  cvt_f32_to_f16_kernel<<<1024, 256, 0, stream>>>(qkv_w, qwh, 3 * DIMC * DIMC / 4);
  cvt_f32_to_f16_kernel<<<512, 256, 0, stream>>>(proj_w, pwh, DIMC * DIMC / 4);

  // QKV: grid = (3072/64 cols, 16384/256 rows)
  qkv_rope_kernel<<<dim3(48, 64), 256, 0, stream>>>(xh, qwh, qkv_b, qbuf, kbuf, vt);

  attn_kernel<<<BATCH * NHEADS * (NTOK / 128), 256, 0, stream>>>(qbuf, kbuf, vt, ao);

  // proj: grid = (1024/64 cols, 16384/256 rows)
  proj_kernel<<<dim3(16, 64), 256, 0, stream>>>(ao, pwh, proj_b, out);
}
